// MDPPInitEmbedding_34548716929628
// MI455X (gfx1250) — matrix operand order repaired and checked by reading, not compiled, not run on hardware
//
#include <hip/hip_runtime.h>

typedef __attribute__((ext_vector_type(16))) _Float16 v16h;
typedef __attribute__((ext_vector_type(8)))  float    v8f;
typedef __attribute__((ext_vector_type(4)))  float    v4f;

#define B_DIM 16
#define N_DIM 2048
#define E_DIM 256
#define K_DIM 512
#define ROWS  (B_DIM * N_DIM)   // 32768

// Async copy: 16 bytes global -> LDS per lane. VDST VGPR holds the LDS byte
// offset, VADDR pair holds the 64-bit global address (ISA 15.18.3 op 98).
#define ASYNC_CP_B128(ldsoff, gaddr)                                          \
    asm volatile("global_load_async_to_lds_b128 %0, %1, off"                  \
                 :: "v"(ldsoff), "v"(gaddr) : "memory")

// Low 32 bits of a generic LDS pointer == LDS byte offset (flat LDS aperture).
__device__ __forceinline__ unsigned lds_off(const void* p) {
    return (unsigned)(unsigned long long)p;
}

// ---------------------------------------------------------------------------
// Kernel 1: min_dist[b,j] = sqrt( min_{i: probe[b,i]} ||locs[b,i]-locs[b,j]||^2 )
// LDS-tiled N-body min. sqrt is monotone -> one sqrt per node, not per pair.
// ---------------------------------------------------------------------------
__global__ void k_min_dist(const float* __restrict__ locs,
                           const unsigned char* __restrict__ probe,
                           float* __restrict__ md) {
    __shared__ float cx[256];
    __shared__ float cy[256];
    __shared__ int   cp[256];
    const int b   = blockIdx.y;
    const int j   = blockIdx.x * 256 + threadIdx.x;
    const float xj = locs[(b * N_DIM + j) * 2 + 0];
    const float yj = locs[(b * N_DIM + j) * 2 + 1];
    float m = 3.0e38f;
    for (int t0 = 0; t0 < N_DIM; t0 += 256) {
        const int i = t0 + threadIdx.x;
        cx[threadIdx.x] = locs[(b * N_DIM + i) * 2 + 0];
        cy[threadIdx.x] = locs[(b * N_DIM + i) * 2 + 1];
        cp[threadIdx.x] = probe[b * N_DIM + i];
        __syncthreads();
        #pragma unroll 8
        for (int i2 = 0; i2 < 256; ++i2) {
            const float dx = xj - cx[i2];
            const float dy = yj - cy[i2];
            const float d2 = dx * dx + dy * dy;
            m = fminf(m, cp[i2] ? d2 : 3.0e38f);
        }
        __syncthreads();
    }
    md[b * N_DIM + j] = sqrtf(fmaxf(m, 0.0f));
}

// ---------------------------------------------------------------------------
// Kernel 2: coefficient tables s0/s1/s2 [512]:
//   cat[r,K] = mul0*s0[K] + mul1*s1[K] + s2[K]
//   (K<256: mul0=x, mul1=y ; K>=256: mul0=min_dist, s1=0)
// ---------------------------------------------------------------------------
__global__ void k_pack_s(const float* __restrict__ Wn, const float* __restrict__ bn,
                         const float* __restrict__ Wd, const float* __restrict__ bd,
                         float* __restrict__ s) {
    const int K = blockIdx.x * 256 + threadIdx.x;
    if (K >= K_DIM) return;
    float s0, s1, s2;
    if (K < E_DIM) { s0 = Wn[K]; s1 = Wn[E_DIM + K]; s2 = bn[K]; }
    else           { const int k2 = K - E_DIM; s0 = Wd[k2]; s1 = 0.0f; s2 = bd[k2]; }
    s[K] = s0; s[K_DIM + K] = s1; s[2 * K_DIM + K] = s2;
}

// ---------------------------------------------------------------------------
// Kernel 3: pre-swizzle W_out (f32 [512,256]) into f16 B-fragments in the
// CDNA5 wave32 32x16 B layout: element j of lane's v16h holds
//   K = 32*kk + (lane&16)/2 + (j&7) + 2*(j&8) ,  N(col) = 16*nt + (lane&15)
// Fragment index: ((kk*16 + nt)*32 + lane)*16 + j  (32B per lane, aligned)
// ---------------------------------------------------------------------------
__global__ void k_pack_b(const float* __restrict__ Wout, _Float16* __restrict__ bfrag) {
    const int idx  = blockIdx.x * 256 + threadIdx.x;   // 131072 total
    const int j    = idx & 15;
    const int lane = (idx >> 4) & 31;
    const int nt   = (idx >> 9) & 15;
    const int kk   = idx >> 13;
    const int K    = 32 * kk + ((lane & 16) >> 1) + (j & 7) + ((j & 8) << 1);
    const int col  = 16 * nt + (lane & 15);
    bfrag[idx] = (_Float16)Wout[K * E_DIM + col];
}

// ---------------------------------------------------------------------------
// Kernel 4: fused WMMA GEMM, async-to-LDS double-buffered B stage.
// Block = 256 thr (8 waves) -> 128 rows x 128 cols. All 8 waves share the
// same 8 B-tiles per k-step (8 KB), so the block DMAs them ONCE into LDS
// with GLOBAL_LOAD_ASYNC_TO_LDS_B128 (2 async instrs per wave per batch),
// prefetching batch kk+1 while batch kk feeds the 8 WMMAs.
// ---------------------------------------------------------------------------
__global__ void k_gemm(const float* __restrict__ locs, const float* __restrict__ md,
                       const float* __restrict__ s, const _Float16* __restrict__ bfragG,
                       const float* __restrict__ b_out, float* __restrict__ out) {
    __shared__ float    ls[3 * K_DIM];          // 6 KB coefficient tables
    __shared__ float    rx[128], ry[128], rmd[128];
    __shared__ _Float16 lB[2][8][32][16];       // 2 x 8KB B-fragment buffers

    const int tid = threadIdx.x;
    for (int i = tid; i < 3 * K_DIM; i += 256) ls[i] = s[i];
    const int rblk = blockIdx.x * 128;
    if (tid < 128) {
        const int r = rblk + tid;
        rx[tid]  = locs[2 * r];
        ry[tid]  = locs[2 * r + 1];
        rmd[tid] = md[r];
    }

    const int ntBase = blockIdx.y * 8;            // first of 8 n-tiles

    // Per-thread async-copy slots: chunk c in [0,512): t=c>>6, ln=(c&63)>>1,
    // half=(c&1). Source byte = kk*16384 + ((ntBase+t)*32+ln)*32 + half*16.
    const int c0 = tid, c1 = 256 + tid;
    const int t0c = c0 >> 6, ln0 = (c0 & 63) >> 1, h0 = c0 & 1;
    const int t1c = c1 >> 6, ln1 = (c1 & 63) >> 1, h1 = c1 & 1;
    const size_t g0 = (size_t)((ntBase + t0c) * 32 + ln0) * 32 + h0 * 16;
    const size_t g1 = (size_t)((ntBase + t1c) * 32 + ln1) * 32 + h1 * 16;
    const unsigned l0a = lds_off(&lB[0][t0c][ln0][h0 * 8]);
    const unsigned l0b = lds_off(&lB[1][t0c][ln0][h0 * 8]);
    const unsigned l1a = lds_off(&lB[0][t1c][ln1][h1 * 8]);
    const unsigned l1b = lds_off(&lB[1][t1c][ln1][h1 * 8]);
    const char* gbase = (const char*)bfragG;

    // prologue: DMA batch kk=0 into buffer 0
    ASYNC_CP_B128(l0a, gbase + g0);
    ASYNC_CP_B128(l1a, gbase + g1);

    __syncthreads();   // rx/ry/rmd/ls staged

    const int wave = tid >> 5;
    const int lane = tid & 31;
    const int lr   = wave * 16 + (lane & 15);     // row feeding this lane's A frag
    const float x = rx[lr], y = ry[lr], mdv = rmd[lr];
    const int kofs = (lane & 16) ? 8 : 0;

    v8f acc[8] = {};

    for (int kk = 0; kk < 16; ++kk) {
        const int cur = kk & 1;

        // own DMA batch kk has landed in LDS
        asm volatile("s_wait_asynccnt 0x0" ::: "memory");
        __syncthreads();                          // publish to all waves

        // prefetch batch kk+1 into the other buffer (overlaps compute below)
        if (kk + 1 < 16) {
            const size_t ko = (size_t)(kk + 1) * 16384;
            ASYNC_CP_B128(cur ? l0a : l0b, gbase + ko + g0);
            ASYNC_CP_B128(cur ? l1a : l1b, gbase + ko + g1);
        }

        // ---- B fragments from LDS (ds_load_b128 pairs) ----
        v16h bf[8];
        #pragma unroll
        for (int t = 0; t < 8; ++t) {
            bf[t] = *(const v16h*)&lB[cur][t][lane][0];
        }

        // ---- build A fragment from coefficient tables ----
        const float mul0 = (kk < 8) ? x : mdv;    // node-half vs dist-half of K
        const float mul1 = y;                     // s1 is zero for kk>=8
        const int kb = kk * 32 + kofs;

        const v4f s0a = *(const v4f*)&ls[kb];
        const v4f s0b = *(const v4f*)&ls[kb + 4];
        const v4f s0c = *(const v4f*)&ls[kb + 16];
        const v4f s0d = *(const v4f*)&ls[kb + 20];
        const v4f s1a = *(const v4f*)&ls[K_DIM + kb];
        const v4f s1b = *(const v4f*)&ls[K_DIM + kb + 4];
        const v4f s1c = *(const v4f*)&ls[K_DIM + kb + 16];
        const v4f s1d = *(const v4f*)&ls[K_DIM + kb + 20];
        const v4f s2a = *(const v4f*)&ls[2 * K_DIM + kb];
        const v4f s2b = *(const v4f*)&ls[2 * K_DIM + kb + 4];
        const v4f s2c = *(const v4f*)&ls[2 * K_DIM + kb + 16];
        const v4f s2d = *(const v4f*)&ls[2 * K_DIM + kb + 20];

        v16h a;
        #pragma unroll
        for (int q = 0; q < 4; ++q) {
            a[q]      = (_Float16)(mul0 * s0a[q] + mul1 * s1a[q] + s2a[q]);
            a[4 + q]  = (_Float16)(mul0 * s0b[q] + mul1 * s1b[q] + s2b[q]);
            a[8 + q]  = (_Float16)(mul0 * s0c[q] + mul1 * s1c[q] + s2c[q]);
            a[12 + q] = (_Float16)(mul0 * s0d[q] + mul1 * s1d[q] + s2d[q]);
        }

        // ---- 8 WMMAs on this k-step ----
        #pragma unroll
        for (int t = 0; t < 8; ++t) {
            acc[t] = __builtin_amdgcn_wmma_f32_16x16x32_f16(
                /*neg_a=*/false, a, /*neg_b=*/false, bf[t],
                /*c_mod=*/(short)0, acc[t], /*reuse_a=*/false, /*reuse_b=*/false);
        }

        // all waves done reading lB[cur] before it is DMA-overwritten (kk+2)
        __syncthreads();
    }

    // D layout: VGPR v -> M = v (lanes 0-15) / v+8 (lanes 16-31); N = lane&15
    const int rbase = rblk + wave * 16 + ((lane & 16) ? 8 : 0);
    #pragma unroll
    for (int t = 0; t < 8; ++t) {
        const int e = (ntBase + t) * 16 + (lane & 15);
        const float bo = b_out[e];
        #pragma unroll
        for (int v = 0; v < 8; ++v) {
            out[(size_t)(rbase + v) * E_DIM + e] = acc[t][v] + bo;
        }
    }
}

// ---------------------------------------------------------------------------
extern "C" void kernel_launch(void* const* d_in, const int* in_sizes, int n_in,
                              void* d_out, int out_size, void* d_ws, size_t ws_size,
                              hipStream_t stream) {
    const float*         locs  = (const float*)d_in[0];
    const unsigned char* probe = (const unsigned char*)d_in[1];
    const float*         W_node = (const float*)d_in[2];
    const float*         b_node = (const float*)d_in[3];
    const float*         W_dist = (const float*)d_in[4];
    const float*         b_dist = (const float*)d_in[5];
    const float*         W_out  = (const float*)d_in[6];
    const float*         b_out  = (const float*)d_in[7];
    float* out = (float*)d_out;

    char* ws = (char*)d_ws;
    float*    md    = (float*)ws;                  // 32768 f32 = 128 KB
    float*    sArr  = (float*)(ws + 131072);       // 3*512 f32 = 6 KB
    _Float16* bfrag = (_Float16*)(ws + 137216);    // 131072 f16 = 256 KB

    // 1) nearest-probe distance
    k_min_dist<<<dim3(N_DIM / 256, B_DIM), 256, 0, stream>>>(locs, probe, md);
    // 2) coefficient tables (same stream orders all dependencies)
    k_pack_s<<<2, 256, 0, stream>>>(W_node, b_node, W_dist, b_dist, sArr);
    // 3) W_out -> f16 WMMA B-fragment layout
    k_pack_b<<<131072 / 256, 256, 0, stream>>>(W_out, bfrag);
    // 4) fused GEMM: 32768 x 512 x 256 via v_wmma_f32_16x16x32_f16
    k_gemm<<<dim3(ROWS / 128, E_DIM / 128), 256, 0, stream>>>(
        locs, md, sArr, bfrag, b_out, out);
}
